// MultiHeadAttention_31164282700240
// MI455X (gfx1250) — matrix operand order repaired and checked by reading, not compiled, not run on hardware
//
#include <hip/hip_runtime.h>

// ---------------------------------------------------------------------------
// MHA forward for MI455X (gfx1250).
// Pipeline: f32->bf16 conversion (once) -> bf16 WMMA GEMM projections (TDM
// tile loads, double buffered) -> causal flash attention (TDM Q/K tiles) ->
// bf16 WMMA output projection (fp32 result).
// ---------------------------------------------------------------------------

#define DMODEL   1024
#define NHEADS   16
#define HEADDIM  64
#define SEQLEN   2048
#define BATCH    4
#define ATTSCALE 0.125f            // 1/sqrt(64)

typedef __attribute__((ext_vector_type(16))) __bf16 v16bf;
typedef __attribute__((ext_vector_type(8)))  float  v8f;
typedef __attribute__((ext_vector_type(4)))  unsigned int u32x4;
typedef __attribute__((ext_vector_type(8)))  int          i32x8;
typedef __attribute__((ext_vector_type(4)))  int          i32x4;

union FragAB { v16bf v; uint4 q[2]; };   // 16 bf16 = 32 B = 8 VGPRs

__device__ __forceinline__ unsigned short f32_bf16(float f) {
  union { float f; unsigned u; } c; c.f = f;
  unsigned u = c.u + 0x7FFFu + ((c.u >> 16) & 1u);   // round-to-nearest-even
  return (unsigned short)(u >> 16);
}

__device__ __forceinline__ v8f wmma_bf16(const FragAB& a, const FragAB& b, v8f c) {
  return __builtin_amdgcn_wmma_f32_16x16x32_bf16(false, a.v, false, b.v,
                                                 (short)0, c, false, false);
}

__device__ __forceinline__ unsigned lds_lo32(const void* p) {
  // generic pointer to LDS: low 32 bits are the LDS byte offset (ISA 10.2)
  return (unsigned)(unsigned long long)p;
}

// Tensor Data Mover: 2-D bf16 tile (tile_w x tile_h elems, row stride in
// elems) from global -> LDS, packed row-major.  D# per ISA ch.8.
// Issue from ONE wave only; completion via s_wait_tensorcnt (per-wave).
// 6-arg builtin form (clang-23 / therock headers): groups 0..3 + ext + cpol.
__device__ __forceinline__ void tdm_load_2d_bf16(const void* gp, unsigned lds,
                                                 int tile_w, int tile_h,
                                                 int stride) {
  unsigned long long ga = (unsigned long long)gp;
  // group0: count=1 | lds_addr | global_addr[56:0] | type=2
  u32x4 g0 = { 1u, lds, (unsigned)ga,
               (unsigned)((ga >> 32) & 0x1FFFFFFu) | (2u << 30) };
  // group1: data_size=1(2B); tensor_dim0=stride; tensor_dim1=1<<20;
  //         tile_dim0/1; tensor_dim0_stride=stride
  i32x8 g1 = { (1 << 16),
               (stride & 0xFFFF) << 16,
               (stride >> 16) & 0xFFFF,
               16 | ((tile_w & 0xFFFF) << 16),
               tile_h & 0xFFFF,
               stride,
               0, 0 };
  i32x4 z4 = { 0, 0, 0, 0 };                 // groups 2/3: unused (2-D tile)
  i32x8 z8 = { 0, 0, 0, 0, 0, 0, 0, 0 };
  __builtin_amdgcn_tensor_load_to_lds(g0, g1, z4, z4, z8, 0);
}

// ---------------------------------------------------------------------------
// f32 -> bf16 bulk conversion (8 elems/thread)
// ---------------------------------------------------------------------------
__global__ __launch_bounds__(256)
void cvt_f32_bf16(const float* __restrict__ src, unsigned short* __restrict__ dst,
                  int n) {
  int i = (blockIdx.x * 256 + threadIdx.x) * 8;
  if (i >= n) return;
  const float4* s = (const float4*)(src + i);
  float4 a = s[0], b = s[1];
  union { unsigned short h[8]; uint4 u; } t;
  t.h[0] = f32_bf16(a.x); t.h[1] = f32_bf16(a.y);
  t.h[2] = f32_bf16(a.z); t.h[3] = f32_bf16(a.w);
  t.h[4] = f32_bf16(b.x); t.h[5] = f32_bf16(b.y);
  t.h[6] = f32_bf16(b.z); t.h[7] = f32_bf16(b.w);
  *(uint4*)(dst + i) = t.u;
}

// ---------------------------------------------------------------------------
// GEMM: Y[M,N] = Xbf16[M,K] * Wbf16[N,K]^T + bias[N]
// Block 256 thr (8 waves); tile 128(M) x 128(N), K-step 32, TDM double buffer.
// Waves 4x2, each owns 32x64 = 2x4 WMMA accumulators.
// ---------------------------------------------------------------------------
template <bool OUT_BF16>
__global__ __launch_bounds__(256)
void gemm_bias_tdm(const unsigned short* __restrict__ X,
                   const unsigned short* __restrict__ W,
                   const float* __restrict__ bias, void* __restrict__ Yv,
                   int M, int N, int K)
{
  __shared__ unsigned short Xs[2][128][32];   // 16 KB (bf16, TDM-filled)
  __shared__ unsigned short Ws[2][128][32];   // 16 KB

  const int tid  = threadIdx.x;
  const int lane = tid & 31;
  const int half = lane >> 4;               // WMMA lane-half
  const int l16  = lane & 15;
  const int wave = tid >> 5;                // 0..7
  const int wm   = wave >> 1;               // 0..3 -> 32-row strip
  const int wn   = wave & 1;                // 0..1 -> 64-col strip
  const int m0   = blockIdx.x * 128;
  const int n0   = blockIdx.y * 128;
  const int NK   = K >> 5;

  v8f acc[2][4] = {};

  if (tid < 32) {     // wave0 drives the Tensor Data Mover
    tdm_load_2d_bf16(X + (size_t)m0 * K, lds_lo32(&Xs[0][0][0]), 32, 128, K);
    tdm_load_2d_bf16(W + (size_t)n0 * K, lds_lo32(&Ws[0][0][0]), 32, 128, K);
  }

  for (int kk = 0; kk < NK; ++kk) {
    const int cur = kk & 1, nxt = cur ^ 1;
    if (tid < 32) {
      if (kk + 1 < NK) {     // prefetch next K-slab into the other buffer
        const size_t ko = (size_t)(kk + 1) * 32;
        tdm_load_2d_bf16(X + (size_t)m0 * K + ko, lds_lo32(&Xs[nxt][0][0]),
                         32, 128, K);
        tdm_load_2d_bf16(W + (size_t)n0 * K + ko, lds_lo32(&Ws[nxt][0][0]),
                         32, 128, K);
        __builtin_amdgcn_s_wait_tensorcnt(2);   // <=2 pending => cur complete
      } else {
        __builtin_amdgcn_s_wait_tensorcnt(0);
      }
    }
    __syncthreads();

    // B fragments: col = l16, K = half*16..+15 contiguous
    FragAB bf[4];
    #pragma unroll
    for (int j = 0; j < 4; ++j) {
      const unsigned short* p = &Ws[cur][wn * 64 + j * 16 + l16][half * 16];
      bf[j].q[0] = *(const uint4*)(p);
      bf[j].q[1] = *(const uint4*)(p + 8);
    }
    // A fragments: row = l16, K = half*16..+15 contiguous
    #pragma unroll
    for (int i = 0; i < 2; ++i) {
      FragAB af;
      const unsigned short* p = &Xs[cur][wm * 32 + i * 16 + l16][0];
      af.q[0] = *(const uint4*)(p + half * 16);
      af.q[1] = *(const uint4*)(p + half * 16 + 8);
      #pragma unroll
      for (int j = 0; j < 4; ++j)
        acc[i][j] = wmma_bf16(af, bf[j], acc[i][j]);
    }
    __syncthreads();   // all reads of buffer `cur` done before it is refilled
  }

  // epilogue: C layout row = v + 8*half, col = l16
  #pragma unroll
  for (int j = 0; j < 4; ++j) {
    int col = n0 + wn * 64 + j * 16 + l16;
    float bv = bias[col];
    #pragma unroll
    for (int i = 0; i < 2; ++i) {
      #pragma unroll
      for (int v = 0; v < 8; ++v) {
        int row = m0 + wm * 32 + i * 16 + v + 8 * half;
        float val = acc[i][j][v] + bv;
        if constexpr (OUT_BF16)
          ((unsigned short*)Yv)[(size_t)row * N + col] = f32_bf16(val);
        else
          ((float*)Yv)[(size_t)row * N + col] = val;
      }
    }
  }
}

// ---------------------------------------------------------------------------
// Causal flash attention on bf16 qh/kh/vh. Grid (S/128, NHEADS, BATCH),
// 256 thr (8 waves, 16 q-rows each). Q and K tiles via TDM; V staged
// transposed for contiguous P*V B-fragments; output bf16.
// ---------------------------------------------------------------------------
__global__ __launch_bounds__(256)
void flash_attn_bf16(const unsigned short* __restrict__ Q,
                     const unsigned short* __restrict__ K,
                     const unsigned short* __restrict__ V,
                     unsigned short* __restrict__ O)
{
  __shared__ unsigned short Qs[128][64];     // 16 KB  Qs[row][d]   (TDM)
  __shared__ unsigned short Ks[64][64];      //  8 KB  Ks[key][d]   (TDM)
  __shared__ unsigned short Vs[64][64];      //  8 KB  Vs[d][key]   (VALU xpose)
  __shared__ unsigned short Ps[8][16][64];   // 16 KB  per-wave P staging

  const int tid  = threadIdx.x;
  const int lane = tid & 31;
  const int half = lane >> 4;
  const int l16  = lane & 15;
  const int wave = tid >> 5;
  const int q0   = blockIdx.x * 128;
  const int head = blockIdx.y;
  const int b    = blockIdx.z;

  const unsigned short* Qb = Q + ((size_t)b * SEQLEN) * DMODEL + head * HEADDIM;
  const unsigned short* Kb = K + ((size_t)b * SEQLEN) * DMODEL + head * HEADDIM;
  const unsigned short* Vb = V + ((size_t)b * SEQLEN) * DMODEL + head * HEADDIM;
  unsigned short*       Ob = O + ((size_t)b * SEQLEN) * DMODEL + head * HEADDIM;

  if (tid < 32)   // Q tile once per block
    tdm_load_2d_bf16(Qb + (size_t)q0 * DMODEL, lds_lo32(&Qs[0][0]),
                     HEADDIM, 128, DMODEL);

  float m_r[8], l_r[8];
  v8f   Oacc[4] = {};
  #pragma unroll
  for (int v = 0; v < 8; ++v) { m_r[v] = -__builtin_inff(); l_r[v] = 0.0f; }

  const int kend = q0 + 128;                 // causal: skip tiles above diag
  const int kvr = tid >> 2, kvc = (tid & 3) * 16;

  for (int k0 = 0; k0 < kend; k0 += 64) {
    __syncthreads();                         // prior reads of Ks/Vs complete
    if (tid < 32)
      tdm_load_2d_bf16(Kb + (size_t)k0 * DMODEL, lds_lo32(&Ks[0][0]),
                       HEADDIM, 64, DMODEL);
    // stage V tile transposed: Vs[d][key]
    {
      const uint4* sv = (const uint4*)(Vb + (size_t)(k0 + kvr) * DMODEL + kvc);
      union { unsigned short h[8]; uint4 u; } t;
      #pragma unroll
      for (int i = 0; i < 2; ++i) {
        t.u = sv[i];
        #pragma unroll
        for (int j = 0; j < 8; ++j)
          Vs[kvc + i * 8 + j][kvr] = t.h[j];
      }
      if (k0 + 64 < kend)                    // global_prefetch_b8 next tile
        __builtin_prefetch(Vb + (size_t)(k0 + 64 + kvr) * DMODEL + kvc, 0, 3);
    }
    if (tid < 32)
      __builtin_amdgcn_s_wait_tensorcnt(0);  // K (and first-iter Q) landed
    __syncthreads();

    // ---- scores S = Q * K^T ----
    v8f s[4] = {};
    #pragma unroll
    for (int dk = 0; dk < 2; ++dk) {
      FragAB af;
      const unsigned short* pa = &Qs[wave * 16 + l16][dk * 32];
      af.q[0] = *(const uint4*)(pa + half * 16);
      af.q[1] = *(const uint4*)(pa + half * 16 + 8);
      #pragma unroll
      for (int j = 0; j < 4; ++j) {
        FragAB bf;
        const unsigned short* pb = &Ks[j * 16 + l16][dk * 32 + half * 16];
        bf.q[0] = *(const uint4*)(pb);
        bf.q[1] = *(const uint4*)(pb + 8);
        s[j] = wmma_bf16(af, bf, s[j]);
      }
    }

    // ---- online softmax (row = v + 8*half within wave strip) ----
    #pragma unroll
    for (int v = 0; v < 8; ++v) {
      const int qg = q0 + wave * 16 + v + 8 * half;
      float rm = -__builtin_inff();
      #pragma unroll
      for (int j = 0; j < 4; ++j) {
        int kc = k0 + j * 16 + l16;
        float val = s[j][v] * ATTSCALE;
        val = (kc > qg) ? -__builtin_inff() : val;   // causal mask
        s[j][v] = val;
        rm = fmaxf(rm, val);
      }
      #pragma unroll
      for (int off = 1; off < 16; off <<= 1)         // reduce within half
        rm = fmaxf(rm, __shfl_xor(rm, off, 32));
      float mn = fmaxf(m_r[v], rm);
      float sc = __expf(m_r[v] - mn);
      float rs = 0.0f;
      #pragma unroll
      for (int j = 0; j < 4; ++j) {
        float p = __expf(s[j][v] - mn);
        s[j][v] = p;
        rs += p;
      }
      #pragma unroll
      for (int off = 1; off < 16; off <<= 1)
        rs += __shfl_xor(rs, off, 32);
      l_r[v] = l_r[v] * sc + rs;
      m_r[v] = mn;
      #pragma unroll
      for (int dt = 0; dt < 4; ++dt)
        Oacc[dt][v] = Oacc[dt][v] * sc;
    }

    // ---- P -> per-wave LDS (C layout -> A layout); same-wave LDS in-order
    #pragma unroll
    for (int j = 0; j < 4; ++j)
      #pragma unroll
      for (int v = 0; v < 8; ++v)
        Ps[wave][v + 8 * half][j * 16 + l16] = f32_bf16(s[j][v]);

    // ---- O += P * V ----
    #pragma unroll
    for (int ks = 0; ks < 2; ++ks) {
      FragAB af;
      const unsigned short* pa = &Ps[wave][l16][ks * 32];
      af.q[0] = *(const uint4*)(pa + half * 16);
      af.q[1] = *(const uint4*)(pa + half * 16 + 8);
      #pragma unroll
      for (int dt = 0; dt < 4; ++dt) {
        FragAB bf;
        const unsigned short* pb = &Vs[dt * 16 + l16][ks * 32 + half * 16];
        bf.q[0] = *(const uint4*)(pb);
        bf.q[1] = *(const uint4*)(pb + 8);
        Oacc[dt] = wmma_bf16(af, bf, Oacc[dt]);
      }
    }
  }

  // ---- normalize + store (bf16) ----
  #pragma unroll
  for (int v = 0; v < 8; ++v) {
    const int qg = q0 + wave * 16 + v + 8 * half;
    float inv = 1.0f / l_r[v];
    #pragma unroll
    for (int dt = 0; dt < 4; ++dt)
      Ob[(size_t)qg * DMODEL + dt * 16 + l16] = f32_bf16(Oacc[dt][v] * inv);
  }
}

// ---------------------------------------------------------------------------
extern "C" void kernel_launch(void* const* d_in, const int* in_sizes, int n_in,
                              void* d_out, int out_size, void* d_ws, size_t ws_size,
                              hipStream_t stream) {
  const float* q  = (const float*)d_in[0];
  const float* k  = (const float*)d_in[1];
  const float* v  = (const float*)d_in[2];
  // d_in[3] = attn_mask (causal) -- computed analytically in-kernel
  const float* Wq = (const float*)d_in[4];
  const float* bq = (const float*)d_in[5];
  const float* Wk = (const float*)d_in[6];
  const float* bk = (const float*)d_in[7];
  const float* Wv = (const float*)d_in[8];
  const float* bv = (const float*)d_in[9];
  const float* Wo = (const float*)d_in[10];
  const float* bo = (const float*)d_in[11];
  float* out = (float*)d_out;

  const int M = BATCH * SEQLEN;                    // 8192
  const size_t mat  = (size_t)M * DMODEL;          // 8.39M elems
  const size_t wmat = (size_t)DMODEL * DMODEL;     // 1.05M elems

  unsigned short* p   = (unsigned short*)d_ws;     // bf16 workspace (~126 MB)
  unsigned short* qb  = p;  p += mat;
  unsigned short* kb  = p;  p += mat;
  unsigned short* vb  = p;  p += mat;
  unsigned short* Wqb = p;  p += wmat;
  unsigned short* Wkb = p;  p += wmat;
  unsigned short* Wvb = p;  p += wmat;
  unsigned short* Wob = p;  p += wmat;
  unsigned short* qh  = p;  p += mat;
  unsigned short* kh  = p;  p += mat;
  unsigned short* vh  = p;  p += mat;
  unsigned short* at  = p;  p += mat;

  dim3 blk(256);
  // one-time f32 -> bf16 conversions (halves all downstream HBM traffic)
  cvt_f32_bf16<<<dim3((unsigned)(mat  / 8 / 256)), blk, 0, stream>>>(q,  qb,  (int)mat);
  cvt_f32_bf16<<<dim3((unsigned)(mat  / 8 / 256)), blk, 0, stream>>>(k,  kb,  (int)mat);
  cvt_f32_bf16<<<dim3((unsigned)(mat  / 8 / 256)), blk, 0, stream>>>(v,  vb,  (int)mat);
  cvt_f32_bf16<<<dim3((unsigned)(wmat / 8 / 256)), blk, 0, stream>>>(Wq, Wqb, (int)wmat);
  cvt_f32_bf16<<<dim3((unsigned)(wmat / 8 / 256)), blk, 0, stream>>>(Wk, Wkb, (int)wmat);
  cvt_f32_bf16<<<dim3((unsigned)(wmat / 8 / 256)), blk, 0, stream>>>(Wv, Wvb, (int)wmat);
  cvt_f32_bf16<<<dim3((unsigned)(wmat / 8 / 256)), blk, 0, stream>>>(Wo, Wob, (int)wmat);

  dim3 ggrid(M / 128, DMODEL / 128);
  gemm_bias_tdm<true ><<<ggrid, blk, 0, stream>>>(qb, Wqb, bq, qh, M, DMODEL, DMODEL);
  gemm_bias_tdm<true ><<<ggrid, blk, 0, stream>>>(kb, Wkb, bk, kh, M, DMODEL, DMODEL);
  gemm_bias_tdm<true ><<<ggrid, blk, 0, stream>>>(vb, Wvb, bv, vh, M, DMODEL, DMODEL);

  dim3 fgrid(SEQLEN / 128, NHEADS, BATCH);
  flash_attn_bf16<<<fgrid, blk, 0, stream>>>(qh, kh, vh, at);

  gemm_bias_tdm<false><<<ggrid, blk, 0, stream>>>(at, Wob, bo, out, M, DMODEL, DMODEL);
}
